// TrulyNeuralMMU_18975165514073
// MI455X (gfx1250) — matrix operand order repaired and checked by reading, not compiled, not run on hardware
//
#include <hip/hip_runtime.h>
#include <hip/hip_bf16.h>

// ---------------------------------------------------------------------------
// TrulyNeuralMMU for gfx1250 (MI455X): flash-attention kernel on
// v_wmma_f32_16x16x32_bf16 with double-buffered async global->LDS staging
// (ASYNCcnt pipeline) and bank-conflict-free LDS tiling.
// wave32, 8 waves/block, 16 query rows/wave, 64-page tiles.
// ---------------------------------------------------------------------------

#define MAX_PAGES 4096
#define ADDR_BITS 48
#define PAGE_BITS 12
#define PNB       (ADDR_BITS - PAGE_BITS)   // 36
#define KEY_DIM   128
#define BATCH     65536
#define VDIM      48                        // 36 mapping bits + 4 perms + pad
#define NTILES    (MAX_PAGES / 64)          // 64 tiles of 64 pages

// padded LDS row strides (bf16 elements); word-stride % 64 != 0 -> no conflicts
#define KT_STRIDE 136                       // 272 B rows   (bank step 68)
#define VT_STRIDE 72                        // 144 B rows   (bank step 36)
#define QS_STRIDE 136                       // 272 B rows

// phase-C LDS offsets (bytes)
#define KT_BYTES  (64 * KT_STRIDE * 2)      // 17408 per buffer
#define VT_BYTES  (VDIM * VT_STRIDE * 2)    // 6912 per buffer
#define VT_BASE   (2 * KT_BYTES)            // 34816
#define PS_BASE   (VT_BASE + 2 * VT_BYTES)  // 48640 (+8*2048 = 65024 <= 64K)

typedef __bf16 bf16_t;
typedef bf16_t v16bf __attribute__((ext_vector_type(16)));
typedef float  v8f   __attribute__((ext_vector_type(8)));

union AFrag {
    v16bf v;
    uint4 q[2];
    bf16_t e[16];
};

static __device__ inline v8f vzero8() {
    v8f z;
#pragma unroll
    for (int i = 0; i < 8; ++i) z[i] = 0.0f;
    return z;
}

static __device__ inline v8f wmma_bf16(v16bf a, v16bf b, v8f c) {
    return __builtin_amdgcn_wmma_f32_16x16x32_bf16(
        false, a, false, b, (short)0, c, false, false);
}

static __device__ inline AFrag lds_frag(const bf16_t* p0, const bf16_t* p1) {
    AFrag f;
    f.q[0] = *(const uint4*)p0;
    f.q[1] = *(const uint4*)p1;
    return f;
}

static __device__ inline float gelu_exact(float x) {
    return 0.5f * x * (1.0f + erff(x * 0.70710678118654752f));
}
static __device__ inline float sigmoidf_(float x) {
    return 1.0f / (1.0f + __expf(-x));
}

// async global->LDS copy of 16 bytes per lane (GVS mode: SGPR base + VGPR off)
#define ASYNC_B128(ldsoff, goff, base)                                         \
    asm volatile("global_load_async_to_lds_b128 %0, %1, %2 offset:0"           \
                 :: "v"(ldsoff), "v"(goff), "s"(base) : "memory")

// ---------------------------------------------------------------------------
// prep kernel: page_keys f32 -> bf16 [4096][128]; V^T table bf16 [48][4096]
// vt[d][p] = d<36 ? pmap[p][36+d] : d<40 ? pperm[p][d-36] : 0
// ---------------------------------------------------------------------------
__global__ __launch_bounds__(256)
void mmu_prep_kernel(const float* __restrict__ pkey,
                     const float* __restrict__ pmap,
                     const float* __restrict__ pperm,
                     bf16_t* __restrict__ kbf,
                     bf16_t* __restrict__ vt)
{
    const int t = blockIdx.x * 256 + threadIdx.x;   // 131072 threads
    {
        const float4 f = *(const float4*)(pkey + (size_t)t * 4);
        union { bf16_t h[4]; uint2 u; } pk;
        pk.h[0] = (bf16_t)f.x; pk.h[1] = (bf16_t)f.y;
        pk.h[2] = (bf16_t)f.z; pk.h[3] = (bf16_t)f.w;
        *(uint2*)(kbf + (size_t)t * 4) = pk.u;
    }
    for (int e = t; e < VDIM * MAX_PAGES; e += 131072) {
        int d = e >> 12, p = e & 4095;
        float v;
        if (d < PNB)      v = pmap[(size_t)p * (2 * PNB) + PNB + d];
        else if (d < 40)  v = pperm[(size_t)p * 4 + (d - PNB)];
        else              v = 0.0f;
        vt[e] = (bf16_t)v;
    }
}

// stage one 64-page tile asynchronously into padded LDS rows:
// 4 K-chunks + 2 V-chunks per thread (exactly 6 async instr per wave)
static __device__ inline void stage_async(unsigned ktoff, unsigned vtoff,
                                          int pbase, const bf16_t* kbf,
                                          const bf16_t* vt, int tid)
{
#pragma unroll
    for (int j = 0; j < 4; ++j) {
        unsigned c = (unsigned)(tid + j * 256);     // 1024 chunks of 16 B
        unsigned row = c >> 4, sub = c & 15;        // 16 chunks per 256 B row
        ASYNC_B128(ktoff + row * (KT_STRIDE * 2) + sub * 16u,
                   (unsigned)pbase * 256u + c * 16u, kbf);
    }
#pragma unroll
    for (int j = 0; j < 2; ++j) {
        // 384 chunks; upper half-block duplicates lower chunks (same data)
        unsigned c = (j == 0) ? (unsigned)tid : (unsigned)(256 + (tid & 127));
        unsigned d = c >> 3, sub = c & 7;           // 8 chunks per 128 B row
        ASYNC_B128(vtoff + d * (VT_STRIDE * 2) + sub * 16u,
                   d * 8192u + (unsigned)pbase * 2u + sub * 16u, vt);
    }
}

// synchronous fallback staging (no workspace): vectorized load + convert
static __device__ inline void stage_sync(bf16_t* Kt, bf16_t* Vt, int pbase,
                                         const float* pkey, const float* pmap,
                                         const float* pperm, int tid)
{
#pragma unroll
    for (int j = 0; j < 8; ++j) {
        int e4 = tid + j * 256;                     // 2048 float4 chunks
        const float4 f = *(const float4*)(pkey + (size_t)pbase * 128 + e4 * 4);
        union { bf16_t h[4]; uint2 u; } pk;
        pk.h[0] = (bf16_t)f.x; pk.h[1] = (bf16_t)f.y;
        pk.h[2] = (bf16_t)f.z; pk.h[3] = (bf16_t)f.w;
        int row = e4 >> 5, col = (e4 & 31) * 4;
        *(uint2*)(Kt + row * KT_STRIDE + col) = pk.u;
    }
#pragma unroll
    for (int j = 0; j < 12; ++j) {
        int e = tid + j * 256;                      // 3072 elems
        int d = e >> 6, p = e & 63;
        float v;
        if (d < PNB)      v = pmap[(size_t)(pbase + p) * (2 * PNB) + PNB + d];
        else if (d < 40)  v = pperm[(size_t)(pbase + p) * 4 + (d - PNB)];
        else              v = 0.0f;
        Vt[d * VT_STRIDE + p] = (bf16_t)v;
    }
}

// ---------------------------------------------------------------------------
// LDS layout (phase-overlapped union inside 64 KB):
//   phase A (MLP):  w1t [256][48] @ 0..24576 ; w2t [128][128] @ 24576..57344
//                   hs 8 x [16][32] @ 57344..65536
//   phase B (Q):    qstage 8 x [16][QS_STRIDE] @ 0..34816
//   phase C (attn): Kt[2][64][KT_STRIDE] @ 0..34816
//                   Vt[2][48][VT_STRIDE] @ 34816..48640
//                   Ps 8 x [16][64] @ 48640..65024
// ---------------------------------------------------------------------------
__global__ __launch_bounds__(256)
void mmu_flash_kernel(const float* __restrict__ vab,
                      const float* __restrict__ pmap,
                      const float* __restrict__ pkey,
                      const float* __restrict__ pperm,
                      const float* __restrict__ W1,
                      const float* __restrict__ b1,
                      const float* __restrict__ W2,
                      const float* __restrict__ b2,
                      const float* __restrict__ temp,
                      const bf16_t* __restrict__ kbf,
                      const bf16_t* __restrict__ vt,
                      int use_async,
                      float* __restrict__ out)
{
    __shared__ __align__(16) char smem[65536];

    const int tid  = threadIdx.x;
    const int lane = tid & 31;
    const int wave = tid >> 5;
    const int l15  = lane & 15;
    const int hi   = lane >> 4;
    const int rowbase = blockIdx.x * 128;
    const int grow = rowbase + wave * 16 + l15;

    const float invtemp = 1.0f / fmaxf(fabsf(temp[0]), 0.1f);

    bf16_t* w1t = (bf16_t*)smem;                            // [256][48]
    bf16_t* w2t = (bf16_t*)(smem + 24576);                  // [128][128]
    bf16_t* hs  = (bf16_t*)(smem + 57344) + wave * 512;     // [16][32]
    bf16_t* qst = (bf16_t*)smem + wave * (16 * QS_STRIDE);  // [16][136]
    bf16_t* Ps  = (bf16_t*)(smem + PS_BASE) + wave * 1024;  // [16][64]

    // -------- offset bits pass-through (float4 copies) ----------------------
    for (int idx = tid; idx < 128 * 3; idx += 256) {
        int r = idx / 3, q = idx % 3;
        *(float4*)(out + (size_t)(rowbase + r) * 52 + q * 4) =
            *(const float4*)(vab + (size_t)(rowbase + r) * 48 + q * 4);
    }

    // -------- phase A1: stage W1^T (pad K to 48) and W2^T half 0 ------------
#pragma unroll
    for (int j = 0; j < 9; ++j) {                 // 2304 float4 chunks of W1
        int c = tid + j * 256;
        int k = c >> 6, n4 = c & 63;
        const float4 f = *(const float4*)(W1 + k * 256 + n4 * 4);
        w1t[(n4 * 4 + 0) * 48 + k] = (bf16_t)f.x;
        w1t[(n4 * 4 + 1) * 48 + k] = (bf16_t)f.y;
        w1t[(n4 * 4 + 2) * 48 + k] = (bf16_t)f.z;
        w1t[(n4 * 4 + 3) * 48 + k] = (bf16_t)f.w;
    }
#pragma unroll
    for (int j = 0; j < 12; ++j) {                // zero pad rows k=36..47
        int c = tid + j * 256;
        int n = c / 12, kz = PNB + (c % 12);
        w1t[n * 48 + kz] = (bf16_t)0.0f;
    }
#pragma unroll
    for (int j = 0; j < 16; ++j) {                // 4096 float4 chunks of W2
        int c = tid + j * 256;
        int kl = c >> 5, n4 = c & 31;
        const float4 f = *(const float4*)(W2 + kl * 128 + n4 * 4);
        w2t[(n4 * 4 + 0) * 128 + kl] = (bf16_t)f.x;
        w2t[(n4 * 4 + 1) * 128 + kl] = (bf16_t)f.y;
        w2t[(n4 * 4 + 2) * 128 + kl] = (bf16_t)f.z;
        w2t[(n4 * 4 + 3) * 128 + kl] = (bf16_t)f.w;
    }
    __syncthreads();

    // -------- X A-fragments (16x32 bf16, K 36 -> 64 zero padded) ------------
    AFrag xf[2];
#pragma unroll
    for (int ks2 = 0; ks2 < 2; ++ks2) {
        int k0 = ks2 * 32 + hi * 16;
#pragma unroll
        for (int i = 0; i < 16; ++i) {
            int k = k0 + i;
            float v = (k < PNB) ? vab[(size_t)grow * 48 + PAGE_BITS + k] : 0.0f;
            xf[ks2].e[i] = (bf16_t)v;
        }
    }

    // -------- phase A2: Q = gelu(X@W1+b1)@W2 + b2 ---------------------------
    v8f qc[8];
#pragma unroll
    for (int nt = 0; nt < 8; ++nt) qc[nt] = vzero8();

    for (int half = 0; half < 2; ++half) {
        if (half == 1) {
            __syncthreads();
#pragma unroll
            for (int j = 0; j < 16; ++j) {
                int c = tid + j * 256;
                int kl = c >> 5, n4 = c & 31;
                const float4 f = *(const float4*)(W2 + (128 + kl) * 128 + n4 * 4);
                w2t[(n4 * 4 + 0) * 128 + kl] = (bf16_t)f.x;
                w2t[(n4 * 4 + 1) * 128 + kl] = (bf16_t)f.y;
                w2t[(n4 * 4 + 2) * 128 + kl] = (bf16_t)f.z;
                w2t[(n4 * 4 + 3) * 128 + kl] = (bf16_t)f.w;
            }
            __syncthreads();
        }
#pragma unroll
        for (int ksl = 0; ksl < 4; ++ksl) {
            int ks = half * 4 + ksl;
#pragma unroll
            for (int sub = 0; sub < 2; ++sub) {
                int ht = ks * 2 + sub;
                v8f hc = vzero8();
#pragma unroll
                for (int ks2 = 0; ks2 < 2; ++ks2) {
                    int kb = ks2 * 32 + hi * 16;
                    AFrag bw;
                    if (kb < 48) {
                        const bf16_t* p = w1t + (ht * 16 + l15) * 48 + kb;
                        bw.q[0] = *(const uint4*)p;
                        bw.q[1] = *(const uint4*)(p + 8);
                    } else {
                        bw.q[0] = make_uint4(0, 0, 0, 0);
                        bw.q[1] = make_uint4(0, 0, 0, 0);
                    }
                    hc = wmma_bf16(xf[ks2].v, bw.v, hc);
                }
                float bias = b1[ht * 16 + l15];
#pragma unroll
                for (int r = 0; r < 8; ++r) {
                    float v = gelu_exact(hc[r] + bias);
                    hs[(r + hi * 8) * 32 + sub * 16 + l15] = (bf16_t)v;
                }
            }
            AFrag ha = lds_frag(hs + l15 * 32 + hi * 16,
                                hs + l15 * 32 + hi * 16 + 8);
#pragma unroll
            for (int nt = 0; nt < 8; ++nt) {
                const bf16_t* p = w2t + (nt * 16 + l15) * 128 + ksl * 32 + hi * 16;
                AFrag bw = lds_frag(p, p + 8);
                qc[nt] = wmma_bf16(ha.v, bw.v, qc[nt]);
            }
        }
    }

    // -------- phase B: Q * (1/temp) -> LDS -> A fragments -------------------
    __syncthreads();
#pragma unroll
    for (int nt = 0; nt < 8; ++nt) {
        float bias = b2[nt * 16 + l15];
#pragma unroll
        for (int r = 0; r < 8; ++r) {
            float v = (qc[nt][r] + bias) * invtemp;
            qst[(r + hi * 8) * QS_STRIDE + nt * 16 + l15] = (bf16_t)v;
        }
    }
    AFrag qa[4];
#pragma unroll
    for (int ks = 0; ks < 4; ++ks) {
        const bf16_t* p = qst + l15 * QS_STRIDE + ks * 32 + hi * 16;
        qa[ks] = lds_frag(p, p + 8);
    }
    __syncthreads();   // qstage dead; K/V buffers may now be written

    // -------- phase C: flash attention, double-buffered async pipeline ------
    float mrow[8], lsum[8];
    v8f acc[3];
#pragma unroll
    for (int r = 0; r < 8; ++r) { mrow[r] = -3.0e38f; lsum[r] = 0.0f; }
#pragma unroll
    for (int nt = 0; nt < 3; ++nt) acc[nt] = vzero8();

    const unsigned lds0 = (unsigned)(size_t)(void*)smem;

    // prologue: stage tile 0 into buffer 0
    if (use_async)
        stage_async(lds0, lds0 + VT_BASE, 0, kbf, vt, tid);
    else
        stage_sync((bf16_t*)smem, (bf16_t*)(smem + VT_BASE), 0,
                   pkey, pmap, pperm, tid);

#pragma unroll 2
    for (int it = 0; it < NTILES; ++it) {
        const int cur = it & 1, nxt = cur ^ 1;
        __syncthreads();   // all waves done reading buffer `nxt` (tile it-1)
        if (it + 1 < NTILES) {
            if (use_async)
                stage_async(lds0 + (unsigned)nxt * KT_BYTES,
                            lds0 + VT_BASE + (unsigned)nxt * VT_BYTES,
                            (it + 1) * 64, kbf, vt, tid);
            else
                stage_sync((bf16_t*)(smem + nxt * KT_BYTES),
                           (bf16_t*)(smem + VT_BASE + nxt * VT_BYTES),
                           (it + 1) * 64, pkey, pmap, pperm, tid);
        }
        if (use_async) {
            if (it + 1 < NTILES)
                asm volatile("s_wait_asynccnt 0x6" ::: "memory"); // tile it done
            else
                asm volatile("s_wait_asynccnt 0x0" ::: "memory");
        }
        __syncthreads();   // tile `it` visible to all waves
        const bf16_t* Kt = (const bf16_t*)(smem + cur * KT_BYTES);
        const bf16_t* Vt = (const bf16_t*)(smem + VT_BASE + cur * VT_BYTES);

        // ---- S = Q K^T : 4 col-tiles x 4 k-steps (batch loads, then WMMA)
        v8f s[4];
#pragma unroll
        for (int nt = 0; nt < 4; ++nt) {
            AFrag bk[4];
#pragma unroll
            for (int ks = 0; ks < 4; ++ks) {
                const bf16_t* p = Kt + (nt * 16 + l15) * KT_STRIDE
                                     + ks * 32 + hi * 16;
                bk[ks] = lds_frag(p, p + 8);
            }
            v8f acc_s = vzero8();
#pragma unroll
            for (int ks = 0; ks < 4; ++ks)
                acc_s = wmma_bf16(qa[ks].v, bk[ks].v, acc_s);
            s[nt] = acc_s;
        }

        // ---- online softmax over this 64-column tile
#pragma unroll
        for (int r = 0; r < 8; ++r) {
            float tmax = fmaxf(fmaxf(s[0][r], s[1][r]), fmaxf(s[2][r], s[3][r]));
#pragma unroll
            for (int msk = 1; msk < 16; msk <<= 1)
                tmax = fmaxf(tmax, __shfl_xor(tmax, msk, 32));
            float mnew  = fmaxf(mrow[r], tmax);
            float alpha = __expf(mrow[r] - mnew);
            float psum  = 0.0f;
#pragma unroll
            for (int nt = 0; nt < 4; ++nt) {
                float pv = __expf(s[nt][r] - mnew);
                s[nt][r] = pv;
                psum += pv;
            }
#pragma unroll
            for (int msk = 1; msk < 16; msk <<= 1)
                psum += __shfl_xor(psum, msk, 32);
            lsum[r] = lsum[r] * alpha + psum;
            mrow[r] = mnew;
#pragma unroll
            for (int nt = 0; nt < 3; ++nt) acc[nt][r] *= alpha;
        }

        // ---- acc += P @ V (P via 2 KB per-wave LDS round-trip)
#pragma unroll
        for (int nt4 = 0; nt4 < 4; ++nt4)
#pragma unroll
            for (int r = 0; r < 8; ++r)
                Ps[(r + hi * 8) * 64 + nt4 * 16 + l15] = (bf16_t)s[nt4][r];
        AFrag pa[2];
#pragma unroll
        for (int ks = 0; ks < 2; ++ks) {
            const bf16_t* p = Ps + l15 * 64 + ks * 32 + hi * 16;
            pa[ks] = lds_frag(p, p + 8);
        }
#pragma unroll
        for (int nt = 0; nt < 3; ++nt) {
            AFrag bv[2];
#pragma unroll
            for (int ks = 0; ks < 2; ++ks) {
                const bf16_t* p = Vt + (nt * 16 + l15) * VT_STRIDE
                                     + ks * 32 + hi * 16;
                bv[ks] = lds_frag(p, p + 8);
            }
#pragma unroll
            for (int ks = 0; ks < 2; ++ks)
                acc[nt] = wmma_bf16(pa[ks].v, bv[ks].v, acc[nt]);
        }
    }

    // -------- epilogue: normalize, sigmoid, scatter -------------------------
#pragma unroll
    for (int r = 0; r < 8; ++r) {
        float inv = 1.0f / lsum[r];
        size_t orow = (size_t)(rowbase + wave * 16 + r + hi * 8) * 52;
#pragma unroll
        for (int nt = 0; nt < 3; ++nt) {
            int col = nt * 16 + l15;
            if (col < 40)
                out[orow + PAGE_BITS + col] = sigmoidf_(acc[nt][r] * inv);
        }
    }
}

extern "C" void kernel_launch(void* const* d_in, const int* in_sizes, int n_in,
                              void* d_out, int out_size, void* d_ws, size_t ws_size,
                              hipStream_t stream) {
    (void)in_sizes; (void)n_in; (void)out_size;
    const float* vab   = (const float*)d_in[0];
    const float* pmap  = (const float*)d_in[1];
    const float* pkey  = (const float*)d_in[2];
    const float* pperm = (const float*)d_in[3];
    const float* W1    = (const float*)d_in[4];
    const float* b1    = (const float*)d_in[5];
    const float* W2    = (const float*)d_in[6];
    const float* b2    = (const float*)d_in[7];
    const float* temp  = (const float*)d_in[8];
    float* out = (float*)d_out;

    const size_t kbf_bytes = (size_t)MAX_PAGES * KEY_DIM * 2;   // 1 MB
    const size_t vt_bytes  = (size_t)VDIM * MAX_PAGES * 2;      // 384 KB
    const int use_async = (d_ws != nullptr && ws_size >= kbf_bytes + vt_bytes) ? 1 : 0;

    bf16_t* kbf = (bf16_t*)d_ws;
    bf16_t* vt  = (bf16_t*)((char*)d_ws + kbf_bytes);

    if (use_async) {
        mmu_prep_kernel<<<dim3(512), dim3(256), 0, stream>>>(pkey, pmap, pperm, kbf, vt);
    }
    mmu_flash_kernel<<<dim3(BATCH / 128), dim3(256), 0, stream>>>(
        vab, pmap, pkey, pperm, W1, b1, W2, b2, temp, kbf, vt, use_async, out);
}
